// ContrastiveLoss_48928267436623
// MI455X (gfx1250) — compile-verified, matrix operand order checked by reading, not run on hardware
//
#include <hip/hip_runtime.h>
#include <hip/hip_bf16.h>

#define EPSV 1e-8f
#define DIMV 1024
#define BM 128
#define BN 128
#define BK 32
#define LDK 40   // padded LDS row stride (bf16 units): 80B rows -> conflict-free b128 reads

typedef __attribute__((ext_vector_type(16))) __bf16 v16bf;
typedef __attribute__((ext_vector_type(8)))  float  v8f;

__device__ __forceinline__ float wave_reduce_add(float v) {
    v += __shfl_xor(v, 16);
    v += __shfl_xor(v, 8);
    v += __shfl_xor(v, 4);
    v += __shfl_xor(v, 2);
    v += __shfl_xor(v, 1);
    return v;
}

// ---------------------------------------------------------------------------
// Kernel 1: row norms, positive-pair cosine sim, f32 -> bf16 hi/lo split,
// and rowsum zeroing (keeps the whole launch graph-capture safe).
// ---------------------------------------------------------------------------
__global__ __launch_bounds__(256)
void cl_prep(const float* __restrict__ x, const float* __restrict__ pos,
             const float* __restrict__ neg,
             __bf16* __restrict__ xhi, __bf16* __restrict__ xlo,
             __bf16* __restrict__ nhi, __bf16* __restrict__ nlo,
             float* __restrict__ nx, float* __restrict__ nn,
             float* __restrict__ sim, float* __restrict__ rowsum,
             int bn, int cn, int dim)
{
    const int b = blockIdx.x;
    const int t = threadIdx.x;
    const int lane = t & 31, wave = t >> 5;
    __shared__ float sred[24];

    if (b < bn) {
        const size_t base = (size_t)b * dim;
        float sx = 0.f, sp = 0.f, dp = 0.f;
        for (int k = t; k < dim; k += 256) {
            float xv = x[base + k], pv = pos[base + k];
            sx += xv * xv; sp += pv * pv; dp += xv * pv;
            __bf16 h = (__bf16)xv;
            xhi[base + k] = h;
            xlo[base + k] = (__bf16)(xv - (float)h);
        }
        sx = wave_reduce_add(sx);
        sp = wave_reduce_add(sp);
        dp = wave_reduce_add(dp);
        if (lane == 0) { sred[wave] = sx; sred[8 + wave] = sp; sred[16 + wave] = dp; }
        __syncthreads();
        if (t == 0) {
            float a = 0.f, c = 0.f, d = 0.f;
            for (int w = 0; w < 8; ++w) { a += sred[w]; c += sred[8 + w]; d += sred[16 + w]; }
            float na = sqrtf(a), nc = sqrtf(c);
            nx[b]  = na;
            sim[b] = d / fmaxf(na * nc, EPSV);
            rowsum[b] = 0.f;
        }
    } else {
        const int j = b - bn;
        const size_t base = (size_t)j * dim;
        float s = 0.f;
        for (int k = t; k < dim; k += 256) {
            float v = neg[base + k];
            s += v * v;
            __bf16 h = (__bf16)v;
            nhi[base + k] = h;
            nlo[base + k] = (__bf16)(v - (float)h);
        }
        s = wave_reduce_add(s);
        if (lane == 0) sred[wave] = s;
        __syncthreads();
        if (t == 0) {
            float a = 0.f;
            for (int w = 0; w < 8; ++w) a += sred[w];
            nn[j] = sqrtf(a);
        }
    }
}

// Load one 16-bf16 WMMA operand fragment: lane holds K {kk..kk+7} and {kk+16..kk+23}
// of its row, per the CDNA5 16-bit A/B VGPR layout.
__device__ __forceinline__ v16bf lds_load_frag(const __bf16* p) {
    union { v16bf v; uint4 q[2]; } u;
    u.q[0] = *(const uint4*)(p);
    u.q[1] = *(const uint4*)(p + 16);
    return u.v;
}

// CDNA5 async global->LDS copy: 16B per lane, same 24-bit immediate offset is
// applied to BOTH the LDS destination and the global source address.
#define ASYNC_B128(ldsoff, gaddr, OFFSTR)                                     \
    asm volatile("global_load_async_to_lds_b128 %0, %1, off offset:" OFFSTR   \
                 :: "v"(ldsoff), "v"(gaddr) : "memory")

__device__ __forceinline__ void stage_row64B(uint32_t lds, const __bf16* g) {
    uint64_t ga = (uint64_t)(uintptr_t)g;
    ASYNC_B128(lds, ga, "0");
    ASYNC_B128(lds, ga, "16");
    ASYNC_B128(lds, ga, "32");
    ASYNC_B128(lds, ga, "48");
}

__device__ __forceinline__ void wait_async0() {
    asm volatile("s_wait_asynccnt 0x0" ::: "memory");
}

// ---------------------------------------------------------------------------
// Kernel 2: fused  exp( (x . neg_j) / max(nx_i*nn_j, eps) )  row-sum GEMM.
// 128x128 tile / block, 8 waves each own a 32x64 sub-tile (2x4 fragments).
// Split-bf16: acc += Ahi*Bhi + Ahi*Blo + Alo*Bhi  (~fp32 accuracy).
// Double-buffered ASYNC global->LDS staging overlapped with WMMA compute.
// ---------------------------------------------------------------------------
__global__ __launch_bounds__(256)
void cl_gemm_exp(const __bf16* __restrict__ xhi, const __bf16* __restrict__ xlo,
                 const __bf16* __restrict__ nhi, const __bf16* __restrict__ nlo,
                 const float* __restrict__ nx, const float* __restrict__ nn,
                 float* __restrict__ rowsum, int dim)
{
    __shared__ __bf16 sAhi[2][BM * LDK], sAlo[2][BM * LDK];
    __shared__ __bf16 sBhi[2][BN * LDK], sBlo[2][BN * LDK];

    const int t     = threadIdx.x;
    const int lane  = t & 31;
    const int wave  = t >> 5;
    const int waveM = wave & 3;   // 4 wave-rows * 32 rows
    const int waveN = wave >> 2;  // 2 wave-cols * 64 cols
    const int mbase = blockIdx.y * BM;
    const int nbase = blockIdx.x * BN;

    v8f acc[2][4] = {};

    // Cooperative staging: threads 0-127 carry the hi planes, 128-255 the lo
    // planes; each thread owns one 64B row-slab of A and one of B per K-step.
    const int lrow   = t & 127;
    const bool hiSel = (t < 128);
    const __bf16* Ag = (hiSel ? xhi : xlo) + (size_t)(mbase + lrow) * dim;
    const __bf16* Bg = (hiSel ? nhi : nlo) + (size_t)(nbase + lrow) * dim;

    // Flat shared pointers truncated to 32 bits == raw LDS byte addresses
    // (ISA: LDS aperture maps addr[31:0] directly to LDS).
    const uint32_t ldsA[2] = {
        (uint32_t)(uintptr_t)((hiSel ? &sAhi[0][0] : &sAlo[0][0]) + lrow * LDK),
        (uint32_t)(uintptr_t)((hiSel ? &sAhi[1][0] : &sAlo[1][0]) + lrow * LDK)};
    const uint32_t ldsB[2] = {
        (uint32_t)(uintptr_t)((hiSel ? &sBhi[0][0] : &sBlo[0][0]) + lrow * LDK),
        (uint32_t)(uintptr_t)((hiSel ? &sBhi[1][0] : &sBlo[1][0]) + lrow * LDK)};

    // Prologue: stage tile 0 into buffer 0.
    stage_row64B(ldsA[0], Ag);
    stage_row64B(ldsB[0], Bg);

    for (int k0 = 0; k0 < dim; k0 += BK) {
        const int buf = (k0 / BK) & 1;

        wait_async0();       // my slab of tile k landed in LDS
        __syncthreads();     // everyone's slabs landed; buf^1 readers of k-1 done

        if (k0 + BK < dim) { // stage tile k+1 into the other buffer (overlaps compute)
            stage_row64B(ldsA[buf ^ 1], Ag + k0 + BK);
            stage_row64B(ldsB[buf ^ 1], Bg + k0 + BK);
        }

        const __bf16* cAhi = &sAhi[buf][0];
        const __bf16* cAlo = &sAlo[buf][0];
        const __bf16* cBhi = &sBhi[buf][0];
        const __bf16* cBlo = &sBlo[buf][0];

        const int kk = (lane >= 16) ? 8 : 0;
        v16bf ah[2], al[2], bh[4], bl[4];
#pragma unroll
        for (int mi = 0; mi < 2; ++mi) {
            const int off = (waveM * 32 + mi * 16 + (lane & 15)) * LDK + kk;
            ah[mi] = lds_load_frag(&cAhi[off]);
            al[mi] = lds_load_frag(&cAlo[off]);
        }
#pragma unroll
        for (int ni = 0; ni < 4; ++ni) {
            const int off = (waveN * 64 + ni * 16 + (lane & 15)) * LDK + kk;
            bh[ni] = lds_load_frag(&cBhi[off]);
            bl[ni] = lds_load_frag(&cBlo[off]);
        }
#pragma unroll
        for (int mi = 0; mi < 2; ++mi)
#pragma unroll
            for (int ni = 0; ni < 4; ++ni) {
                acc[mi][ni] = __builtin_amdgcn_wmma_f32_16x16x32_bf16(
                    false, ah[mi], false, bh[ni], (short)0, acc[mi][ni], false, false);
                acc[mi][ni] = __builtin_amdgcn_wmma_f32_16x16x32_bf16(
                    false, ah[mi], false, bl[ni], (short)0, acc[mi][ni], false, false);
                acc[mi][ni] = __builtin_amdgcn_wmma_f32_16x16x32_bf16(
                    false, al[mi], false, bh[ni], (short)0, acc[mi][ni], false, false);
            }
    }

    // Epilogue: scale -> exp -> half-wave column reduce -> atomic row accumulate.
    float nnv[4];
#pragma unroll
    for (int ni = 0; ni < 4; ++ni)
        nnv[ni] = nn[nbase + waveN * 64 + ni * 16 + (lane & 15)];

    const int mrow0 = mbase + waveM * 32 + ((lane >= 16) ? 8 : 0);
#pragma unroll
    for (int mi = 0; mi < 2; ++mi) {
#pragma unroll
        for (int r = 0; r < 8; ++r) {
            const int m = mrow0 + mi * 16 + r;   // C layout: VGPR r = row r (+8 in hi half)
            const float nxv = nx[m];
            float e = 0.f;
#pragma unroll
            for (int ni = 0; ni < 4; ++ni) {
                float s = acc[mi][ni][r] / fmaxf(nxv * nnv[ni], EPSV);
                e += __expf(s);
            }
            // reduce the 16 columns held by this half-wave (masks < 16 keep halves apart)
            e += __shfl_xor(e, 1);
            e += __shfl_xor(e, 2);
            e += __shfl_xor(e, 4);
            e += __shfl_xor(e, 8);
            if ((lane & 15) == 0) atomicAdd(&rowsum[m], e);
        }
    }
}

// ---------------------------------------------------------------------------
// Kernel 3: loss = -(1/bn) * sum_i ( sim_i - log(rowsum_i) )
// ---------------------------------------------------------------------------
__global__ __launch_bounds__(256)
void cl_finalize(const float* __restrict__ sim, const float* __restrict__ rowsum,
                 float* __restrict__ out, int bn)
{
    __shared__ float sred[8];
    float p = 0.f;
    for (int i = threadIdx.x; i < bn; i += 256)
        p += sim[i] - logf(rowsum[i]);
    p = wave_reduce_add(p);
    const int lane = threadIdx.x & 31, wave = threadIdx.x >> 5;
    if (lane == 0) sred[wave] = p;
    __syncthreads();
    if (threadIdx.x == 0) {
        float tot = 0.f;
        for (int w = 0; w < 8; ++w) tot += sred[w];
        out[0] = -tot / (float)bn;
    }
}

extern "C" void kernel_launch(void* const* d_in, const int* in_sizes, int n_in,
                              void* d_out, int out_size, void* d_ws, size_t ws_size,
                              hipStream_t stream)
{
    (void)n_in; (void)out_size; (void)ws_size;
    const int dim = DIMV;
    const int bn  = in_sizes[0] / dim;   // 4096
    const int cn  = in_sizes[2] / dim;   // 8192

    const float* x   = (const float*)d_in[0];
    const float* pos = (const float*)d_in[1];
    const float* neg = (const float*)d_in[2];

    char* w = (char*)d_ws;
    __bf16* xhi = (__bf16*)w; w += (size_t)bn * dim * sizeof(__bf16);
    __bf16* xlo = (__bf16*)w; w += (size_t)bn * dim * sizeof(__bf16);
    __bf16* nhi = (__bf16*)w; w += (size_t)cn * dim * sizeof(__bf16);
    __bf16* nlo = (__bf16*)w; w += (size_t)cn * dim * sizeof(__bf16);
    float* nxv    = (float*)w; w += (size_t)bn * sizeof(float);
    float* nnv    = (float*)w; w += (size_t)cn * sizeof(float);
    float* simv   = (float*)w; w += (size_t)bn * sizeof(float);
    float* rowsum = (float*)w; w += (size_t)bn * sizeof(float);

    cl_prep<<<bn + cn, 256, 0, stream>>>(x, pos, neg, xhi, xlo, nhi, nlo,
                                         nxv, nnv, simv, rowsum, bn, cn, dim);

    dim3 grid(cn / BN, bn / BM);
    cl_gemm_exp<<<grid, 256, 0, stream>>>(xhi, xlo, nhi, nlo, nxv, nnv, rowsum, dim);

    cl_finalize<<<1, 256, 0, stream>>>(simv, rowsum, (float*)d_out, bn);
}
